// Qwen3OmniMoeVisionAttention_29961691857441
// MI455X (gfx1250) — matrix operand order repaired and checked by reading, not compiled, not run on hardware
//
#include <hip/hip_runtime.h>
#include <hip/hip_bf16.h>

// MI455X / gfx1250 implementation: bf16 WMMA (16x16x32) pipeline.
// hidden/w -> bf16, QKV GEMM + bias, in-place RoPE, segmented attention
// (128 segs x 16 heads, S=256, Dh=80 padded to 96), projection GEMM -> f32.

typedef __bf16 bf16_t;
typedef __attribute__((ext_vector_type(16))) __bf16 v16bf;
typedef __attribute__((ext_vector_type(8)))  float  v8f;

#define WMMA_BF16(a, b, c) \
  __builtin_amdgcn_wmma_f32_16x16x32_bf16(false, (a), false, (b), (short)0, (c), false, false)

// A-matrix fragment (16x32 bf16): lanes 0-15 rows M, VGPR0-3 K=0..7(+8*half), VGPR4-7 K=16..23(+8*half)
__device__ inline v16bf frag_a(const bf16_t* lds, int row, int stride, int kBase, int lane) {
  v16bf f;
  unsigned int* fu = reinterpret_cast<unsigned int*>(&f);
  const unsigned short* p = reinterpret_cast<const unsigned short*>(lds) + row * stride;
  const int half = (lane >> 4) & 1;
#pragma unroll
  for (int g = 0; g < 8; ++g) {
    int kk = kBase + ((g < 4) ? (2 * g) : (16 + 2 * (g - 4))) + 8 * half;
    fu[g] = *reinterpret_cast<const unsigned int*>(p + kk);
  }
  return f;
}

// B-matrix fragment (32x16 bf16): lane = column N, VGPR0-3 K=0..7(+8*half), VGPR4-7 K=16..23(+8*half)
__device__ inline v16bf frag_b(const bf16_t* lds, int row, int stride, int kBase, int lane) {
  v16bf f;
  unsigned int* fu = reinterpret_cast<unsigned int*>(&f);
  const unsigned short* p = reinterpret_cast<const unsigned short*>(lds) + row * stride;
  const int half = (lane >> 4) & 1;
#pragma unroll
  for (int g = 0; g < 8; ++g) {
    int kk = kBase + ((g < 4) ? (2 * g) : (16 + 2 * (g - 4))) + 8 * half;
    fu[g] = *reinterpret_cast<const unsigned int*>(p + kk);
  }
  return f;
}

// ---------------------------------------------------------------- converts
__global__ __launch_bounds__(256) void f32_to_bf16_kernel(const float* __restrict__ in,
                                                          bf16_t* __restrict__ out,
                                                          long n) {
  long i = ((long)blockIdx.x * blockDim.x + threadIdx.x) * 4;
  if (i + 3 < n) {
    float4 v = *reinterpret_cast<const float4*>(in + i);
    out[i + 0] = (bf16_t)v.x;
    out[i + 1] = (bf16_t)v.y;
    out[i + 2] = (bf16_t)v.z;
    out[i + 3] = (bf16_t)v.w;
  }
}

// ---------------------------------------------------------------- GEMM
// C[M,N] = A[M,K] * W[N,K]^T + bias ; block tile 128x128, K-tile 32, 8 waves
template <bool OUT_BF16>
__global__ __launch_bounds__(256) void gemm_bf16_kernel(const bf16_t* __restrict__ A,
                                                        const bf16_t* __restrict__ W,
                                                        const float* __restrict__ bias,
                                                        void* __restrict__ outp,
                                                        int K, int N) {
  const int tid  = threadIdx.x;
  const int lane = tid & 31;
  const int wave = tid >> 5;
  const int nr   = lane & 15;
  const int half = (lane >> 4) & 1;
  const int m0 = blockIdx.y * 128;
  const int n0 = blockIdx.x * 128;
  const int wm = (wave & 3) * 32;   // 4 wave-rows of 32
  const int wn = (wave >> 2) * 64;  // 2 wave-cols of 64

  __shared__ bf16_t As[128 * 36];
  __shared__ bf16_t Ws[128 * 36];

  v8f acc[2][4];
#pragma unroll
  for (int mt = 0; mt < 2; ++mt)
#pragma unroll
    for (int nt = 0; nt < 4; ++nt)
      acc[mt][nt] = (v8f){0.f, 0.f, 0.f, 0.f, 0.f, 0.f, 0.f, 0.f};

  const int lrow  = tid >> 1;
  const int lhalf = tid & 1;

  for (int kt = 0; kt < K; kt += 32) {
    const uint4* ga = reinterpret_cast<const uint4*>(&A[(size_t)(m0 + lrow) * K + kt + lhalf * 16]);
    uint4 a0 = ga[0], a1 = ga[1];
    const uint4* gw = reinterpret_cast<const uint4*>(&W[(size_t)(n0 + lrow) * K + kt + lhalf * 16]);
    uint4 w0 = gw[0], w1 = gw[1];

    __syncthreads();  // previous compute done before overwriting LDS
    unsigned int av[8] = {a0.x, a0.y, a0.z, a0.w, a1.x, a1.y, a1.z, a1.w};
    unsigned int wv[8] = {w0.x, w0.y, w0.z, w0.w, w1.x, w1.y, w1.z, w1.w};
    unsigned int* asu = reinterpret_cast<unsigned int*>(As);
    unsigned int* wsu = reinterpret_cast<unsigned int*>(Ws);
    const int base = lrow * 18 + lhalf * 8;
#pragma unroll
    for (int j = 0; j < 8; ++j) {
      asu[base + j] = av[j];
      wsu[base + j] = wv[j];
    }
    __syncthreads();

    v16bf af[2], bfb[4];
#pragma unroll
    for (int mt = 0; mt < 2; ++mt) af[mt] = frag_a(As, wm + mt * 16 + nr, 36, 0, lane);
#pragma unroll
    for (int nt = 0; nt < 4; ++nt) bfb[nt] = frag_b(Ws, wn + nt * 16 + nr, 36, 0, lane);
#pragma unroll
    for (int mt = 0; mt < 2; ++mt)
#pragma unroll
      for (int nt = 0; nt < 4; ++nt)
        acc[mt][nt] = WMMA_BF16(af[mt], bfb[nt], acc[mt][nt]);
  }

#pragma unroll
  for (int mt = 0; mt < 2; ++mt)
#pragma unroll
    for (int nt = 0; nt < 4; ++nt)
#pragma unroll
      for (int r = 0; r < 8; ++r) {
        int row = m0 + wm + mt * 16 + r + 8 * half;
        int col = n0 + wn + nt * 16 + nr;
        float v = acc[mt][nt][r] + bias[col];
        if (OUT_BF16)
          ((bf16_t*)outp)[(size_t)row * N + col] = (bf16_t)v;
        else
          ((float*)outp)[(size_t)row * N + col] = v;
      }
}

// ---------------------------------------------------------------- RoPE (in-place on qkv, q & k planes)
__global__ __launch_bounds__(256) void rope_kernel(bf16_t* __restrict__ qkv,
                                                   const float* __restrict__ cosT,
                                                   const float* __restrict__ sinT) {
  int idx = blockIdx.x * blockDim.x + threadIdx.x;  // T*H*40*2 threads
  int which = idx & 1;          // 0 = q, 1 = k
  int rest = idx >> 1;
  int d = rest % 40;
  int th = rest / 40;
  int h = th & 15;
  int t = th >> 4;
  size_t base = (size_t)t * 3840 + which * 1280 + h * 80;
  float x1 = (float)qkv[base + d];
  float x2 = (float)qkv[base + d + 40];
  float c1 = cosT[t * 80 + d], s1 = sinT[t * 80 + d];
  float c2 = cosT[t * 80 + d + 40], s2 = sinT[t * 80 + d + 40];
  qkv[base + d]      = (bf16_t)(x1 * c1 - x2 * s1);
  qkv[base + d + 40] = (bf16_t)(x2 * c2 + x1 * s2);
}

// ---------------------------------------------------------------- attention
// One block per (segment, head). 8 query-chunks of 32 rows; keys streamed in
// 4 tiles of 64. Dh=80 padded to 96. Scores f32 in LDS, converted in-place to
// bf16 probabilities for the P*V WMMA pass (V stored transposed in LDS).
__global__ __launch_bounds__(256) void attn_kernel(const bf16_t* __restrict__ qkv,
                                                   bf16_t* __restrict__ out) {
  const int seg  = blockIdx.x >> 4;
  const int head = blockIdx.x & 15;
  const int tid  = threadIdx.x;
  const int lane = tid & 31;
  const int wave = tid >> 5;
  const int nr   = lane & 15;
  const int half = (lane >> 4) & 1;

  __shared__ bf16_t Qs[32 * 100];     // 6.4 KB
  __shared__ float  Sf[32 * 257];     // 32.9 KB (reused as bf16 P, stride 514)
  __shared__ bf16_t KVs[96 * 68];     // 13.1 KB (K tiles: 64x100; Vt tiles: 96x68)
  __shared__ float  red[512];

  const size_t rowStride = 3840;
  const size_t segBase = (size_t)seg * 256 * rowStride;
  const int qcol = head * 80;
  const int kcol = 1280 + head * 80;
  const int vcol = 2560 + head * 80;
  const float scale = 0.1118033988749895f;  // 1/sqrt(80)
  bf16_t* Sb = reinterpret_cast<bf16_t*>(Sf);

  for (int chunk = 0; chunk < 8; ++chunk) {
    // ---- load Q chunk (32 x 96, zero-padded beyond 80)
    for (int idx = tid; idx < 32 * 96; idx += 256) {
      int r = idx / 96, d = idx % 96;
      float v = 0.f;
      if (d < 80) v = (float)qkv[segBase + (size_t)(chunk * 32 + r) * rowStride + qcol + d];
      Qs[r * 100 + d] = (bf16_t)v;
    }
    // ---- S = Q*K^T (stream K in tiles of 64 keys)
    for (int kt = 0; kt < 4; ++kt) {
      __syncthreads();
      for (int idx = tid; idx < 64 * 96; idx += 256) {
        int r = idx / 96, d = idx % 96;
        float v = 0.f;
        if (d < 80) v = (float)qkv[segBase + (size_t)(kt * 64 + r) * rowStride + kcol + d];
        KVs[r * 100 + d] = (bf16_t)v;
      }
      __syncthreads();
      const int mt = wave >> 2;  // 0..1
      const int nt = wave & 3;   // 0..3
      v8f acc = (v8f){0.f, 0.f, 0.f, 0.f, 0.f, 0.f, 0.f, 0.f};
#pragma unroll
      for (int kd = 0; kd < 96; kd += 32) {
        v16bf a = frag_a(Qs, mt * 16 + nr, 100, kd, lane);
        v16bf b = frag_b(KVs, nt * 16 + nr, 100, kd, lane);
        acc = WMMA_BF16(a, b, acc);
      }
#pragma unroll
      for (int r = 0; r < 8; ++r) {
        int srow = mt * 16 + r + 8 * half;
        int scol = kt * 64 + nt * 16 + nr;
        Sf[srow * 257 + scol] = acc[r];
      }
    }
    __syncthreads();
    // ---- softmax (8 threads per row)
    const int row = tid >> 3, sub = tid & 7;
    float m = -3.0e38f;
    for (int j = 0; j < 32; ++j) m = fmaxf(m, Sf[row * 257 + sub * 32 + j] * scale);
    red[row * 8 + sub] = m;
    __syncthreads();
    float rm = red[row * 8];
#pragma unroll
    for (int j = 1; j < 8; ++j) rm = fmaxf(rm, red[row * 8 + j]);
    float s = 0.f;
    for (int j = 0; j < 32; ++j) {
      float e = __expf(Sf[row * 257 + sub * 32 + j] * scale - rm);
      Sf[row * 257 + sub * 32 + j] = e;
      s += e;
    }
    red[256 + row * 8 + sub] = s;
    __syncthreads();
    float rs = 0.f;
#pragma unroll
    for (int j = 0; j < 8; ++j) rs += red[256 + row * 8 + j];
    const float inv = 1.f / rs;
    float ev[32];
    for (int j = 0; j < 32; ++j) ev[j] = Sf[row * 257 + sub * 32 + j];
    __syncthreads();  // all f32 reads done before bf16 in-place overwrite
    for (int j = 0; j < 32; ++j) Sb[row * 514 + sub * 32 + j] = (bf16_t)(ev[j] * inv);

    // ---- O = P*V (stream V in tiles of 64 keys, transposed into LDS)
    v8f oa = (v8f){0.f, 0.f, 0.f, 0.f, 0.f, 0.f, 0.f, 0.f};
    v8f ob = (v8f){0.f, 0.f, 0.f, 0.f, 0.f, 0.f, 0.f, 0.f};
    const int t0 = wave, t1 = wave + 8;  // 12 output tiles (2 x 6)
    const int mt0 = t0 / 6, nt0 = t0 % 6;
    const int mt1 = t1 / 6, nt1 = t1 % 6;
    for (int kt = 0; kt < 4; ++kt) {
      __syncthreads();  // P conversion / prior Vt use complete
      for (int idx = tid; idx < 64 * 96; idx += 256) {
        int s2 = idx / 96, d = idx % 96;
        float v = 0.f;
        if (d < 80) v = (float)qkv[segBase + (size_t)(kt * 64 + s2) * rowStride + vcol + d];
        KVs[d * 68 + s2] = (bf16_t)v;  // transposed: Vt[d][s]
      }
      __syncthreads();
#pragma unroll
      for (int kb = 0; kb < 64; kb += 32) {
        v16bf a0 = frag_a(Sb, mt0 * 16 + nr, 514, kt * 64 + kb, lane);
        v16bf b0 = frag_b(KVs, nt0 * 16 + nr, 68, kb, lane);
        oa = WMMA_BF16(a0, b0, oa);
        if (t1 < 12) {
          v16bf a1 = frag_a(Sb, mt1 * 16 + nr, 514, kt * 64 + kb, lane);
          v16bf b1 = frag_b(KVs, nt1 * 16 + nr, 68, kb, lane);
          ob = WMMA_BF16(a1, b1, ob);
        }
      }
    }
    // ---- store O (d < 80 only)
#pragma unroll
    for (int r = 0; r < 8; ++r) {
      int qrow = chunk * 32 + mt0 * 16 + r + 8 * half;
      int d = nt0 * 16 + nr;
      if (d < 80)
        out[(size_t)(seg * 256 + qrow) * 1280 + head * 80 + d] = (bf16_t)oa[r];
    }
    if (t1 < 12) {
#pragma unroll
      for (int r = 0; r < 8; ++r) {
        int qrow = chunk * 32 + mt1 * 16 + r + 8 * half;
        int d = nt1 * 16 + nr;
        if (d < 80)
          out[(size_t)(seg * 256 + qrow) * 1280 + head * 80 + d] = (bf16_t)ob[r];
      }
    }
  }
}

// ---------------------------------------------------------------- launch
extern "C" void kernel_launch(void* const* d_in, const int* in_sizes, int n_in,
                              void* d_out, int out_size, void* d_ws, size_t ws_size,
                              hipStream_t stream) {
  const float* hidden = (const float*)d_in[0];
  const float* cosT   = (const float*)d_in[1];
  const float* sinT   = (const float*)d_in[2];
  const float* w_qkv  = (const float*)d_in[3];
  const float* b_qkv  = (const float*)d_in[4];
  const float* w_proj = (const float*)d_in[5];
  const float* b_proj = (const float*)d_in[6];
  // d_in[7] = cu_seqlens: segments are uniform (S=256), not needed.

  char* ws = (char*)d_ws;
  bf16_t* hb    = (bf16_t*)(ws + 0);             // 32768x1280 bf16   (83,886,080 B)
  bf16_t* wqb   = (bf16_t*)(ws + 83886080ll);    // 3840x1280 bf16    ( 9,830,400 B)
  bf16_t* wpb   = (bf16_t*)(ws + 93716480ll);    // 1280x1280 bf16    ( 3,276,800 B)
  bf16_t* qkvb  = (bf16_t*)(ws + 96993280ll);    // 32768x3840 bf16   (251,658,240 B)
  bf16_t* attnb = (bf16_t*)(ws + 348651520ll);   // 32768x1280 bf16   (83,886,080 B)

  f32_to_bf16_kernel<<<40960, 256, 0, stream>>>(hidden, hb, 41943040ll);
  f32_to_bf16_kernel<<<4800, 256, 0, stream>>>(w_qkv, wqb, 4915200ll);
  f32_to_bf16_kernel<<<1600, 256, 0, stream>>>(w_proj, wpb, 1638400ll);

  dim3 g1(30, 256);
  gemm_bf16_kernel<true><<<g1, 256, 0, stream>>>(hb, wqb, b_qkv, (void*)qkvb, 1280, 3840);

  rope_kernel<<<163840, 256, 0, stream>>>(qkvb, cosT, sinT);

  attn_kernel<<<2048, 256, 0, stream>>>(qkvb, attnb);

  dim3 g2(10, 256);
  gemm_bf16_kernel<false><<<g2, 256, 0, stream>>>(attnb, wpb, b_proj, d_out, 1280, 1280);
}
